// GLTransformerBlock_61469571940759
// MI455X (gfx1250) — matrix operand order repaired and checked by reading, not compiled, not run on hardware
//
#include <hip/hip_runtime.h>
#include <hip/hip_bf16.h>
#include <math.h>

// ---------------------------------------------------------------------------
// GL-Transformer block for MI455X (gfx1250, wave32, WMMA).
// bf16 operands + f32 accumulation via v_wmma_f32_16x16x32_bf16.
// All fragment loads are 16B-vectorized; weights pre-packed to lane layout.
// ---------------------------------------------------------------------------

typedef __bf16 bf16_t;
typedef __attribute__((ext_vector_type(16))) __bf16 v16bf;
typedef __attribute__((ext_vector_type(8)))  __bf16 v8bf;
typedef __attribute__((ext_vector_type(8)))  float  v8f;

#define B_          2
#define C_          192
#define HEADS_      6
#define HD_         32
#define NL_         64
#define NWB_        2048          // windows per batch (8*16*16)
#define NG_         2048
#define L_          131072
#define HID_        768
#define ROWS_LOCAL_ 262144        // B_ * L_  (windowed rows)
#define ROWS_GLOB_  4096          // B_ * NG_
#define ROWS_ALL_   266240
#define SCALE_      0.17677669529663687f

__device__ __forceinline__ int lane_id() { return threadIdx.x & 31; }

// ---- WMMA fragment loaders (contiguous k order, shared by A and B) -----
// A fragment 16x32 (MxK) from row-major src[m*ld+k]:
//   lanes 0-15: M=lane, K=0..15; lanes 16-31: M=lane-16, K=16..31
//   -> per lane: two contiguous 8x bf16 runs => two 16B loads.
__device__ __forceinline__ v16bf load_a_rowmajor(const bf16_t* src, int ld) {
    const int l  = lane_id();
    const int m  = l & 15;
    const int kb = (l < 16) ? 0 : 16;
    const v8bf lo = *(const v8bf*)(src + m * ld + kb);
    const v8bf hi = *(const v8bf*)(src + m * ld + kb + 8);
    v16bf a;
#pragma unroll
    for (int j = 0; j < 8; ++j) { a[j] = lo[j]; a[8 + j] = hi[j]; }
    return a;
}

// B fragment 32x16 (KxN) where the 16 K-values of each lane are contiguous in
// memory: element(k,n) = src[n*ld + k] (i.e. N-major / transposed source).
// lanes 0-15: N=lane, K=0..15; lanes 16-31: N=lane-16, K=16..31.
__device__ __forceinline__ v16bf load_b_contig(const bf16_t* src, int ld) {
    const int l  = lane_id();
    const int n  = l & 15;
    const int kb = (l < 16) ? 0 : 16;
    const v8bf lo = *(const v8bf*)(src + n * ld + kb);
    const v8bf hi = *(const v8bf*)(src + n * ld + kb + 8);
    v16bf b;
#pragma unroll
    for (int j = 0; j < 8; ++j) { b[j] = lo[j]; b[8 + j] = hi[j]; }
    return b;
}

// B fragment from pre-packed weights: 512 bf16 per (kc,tn) fragment, 16
// contiguous values per lane -> two 16B loads.
__device__ __forceinline__ v16bf load_b_packed(const bf16_t* p) {
    const int l = lane_id();
    const v8bf lo = *(const v8bf*)(p + l * 16);
    const v8bf hi = *(const v8bf*)(p + l * 16 + 8);
    v16bf b;
#pragma unroll
    for (int j = 0; j < 8; ++j) { b[j] = lo[j]; b[8 + j] = hi[j]; }
    return b;
}

__device__ __forceinline__ v8f wmma_bf16(v16bf a, v16bf b, v8f c) {
    return __builtin_amdgcn_wmma_f32_16x16x32_bf16(
        /*neg_a=*/false, a, /*neg_b=*/false, b,
        /*c_mod=*/(short)0, c, /*reuse_a=*/false, /*reuse_b=*/false);
}
// C/D layout: element r of v8f at lane l -> (M = r + 8*(l>=16), N = l&15).

// ---------------------------------------------------------------------------
// Pack f32 weights W[K,N] into WMMA B-fragment order:
// Bp[((kc*ntiles+tn)*32 + lane)*16 + j] = W[(kc*32 + kb(lane) + j)*N + tn*16 + (lane&15)]
__global__ void k_pack_b(const float* __restrict__ W, bf16_t* __restrict__ Bp,
                         int K, int N) {
    const size_t i = (size_t)blockIdx.x * blockDim.x + threadIdx.x;
    if (i >= (size_t)K * N) return;
    const int j = (int)(i & 15);
    const int l = (int)((i >> 4) & 31);
    const size_t ft = i >> 9;
    const int ntiles = N >> 4;
    const int tn = (int)(ft % ntiles);
    const int kc = (int)(ft / ntiles);
    const int n  = l & 15;
    const int kb = (l < 16) ? 0 : 16;
    Bp[i] = (bf16_t)W[(size_t)(kc * 32 + kb + j) * N + tn * 16 + n];
}

// ---------------------------------------------------------------------------
// LayerNorm (C=192). One wave per row; 6 values per lane; wave32 shfl reduce.
// mode 1: local tokens (B,L,C) -> window-partitioned output rows.
__global__ void k_ln(const float* __restrict__ X, const float* __restrict__ gamma,
                     const float* __restrict__ beta, bf16_t* __restrict__ Out,
                     int nrows, int mode, int out_row_offset) {
    const int row = blockIdx.x * (blockDim.x >> 5) + (threadIdx.x >> 5);
    if (row >= nrows) return;
    const int l = lane_id();
    const float* xr = X + (size_t)row * C_;
    float v[6];
    float s = 0.f;
#pragma unroll
    for (int j = 0; j < 6; ++j) { v[j] = xr[l + 32 * j]; s += v[j]; }
#pragma unroll
    for (int d = 16; d >= 1; d >>= 1) s += __shfl_xor(s, d, 32);
    const float mean = s * (1.0f / C_);
    float var = 0.f;
#pragma unroll
    for (int j = 0; j < 6; ++j) { const float t = v[j] - mean; var += t * t; }
#pragma unroll
    for (int d = 16; d >= 1; d >>= 1) var += __shfl_xor(var, d, 32);
    const float inv = rsqrtf(var * (1.0f / C_) + 1e-5f);

    size_t orow;
    if (mode == 1) {
        const int b = row >> 17;            // L_ = 2^17
        const int t = row & (L_ - 1);
        const int sI = t >> 12;             // H*W = 4096
        const int hI = (t >> 6) & 63;
        const int wI = t & 63;
        const int win = ((sI >> 2) << 8) + ((hI >> 2) << 4) + (wI >> 2);
        const int wr  = ((sI & 3) << 4) + ((hI & 3) << 2) + (wI & 3);
        orow = ((size_t)b * NWB_ + win) * NL_ + wr;
    } else {
        orow = (size_t)out_row_offset + row;
    }
    bf16_t* orp = Out + orow * C_;
#pragma unroll
    for (int j = 0; j < 6; ++j) {
        const int c = l + 32 * j;
        orp[c] = (bf16_t)((v[j] - mean) * inv * gamma[c] + beta[c]);
    }
}

// ---------------------------------------------------------------------------
// WMMA GEMM: Out[M,N] = A[M,K](bf16 row-major) x Bp(packed bf16) + bias.
// One wave computes 16 rows x 64 cols (4 N-tiles) for 4x A-fragment reuse.
// flags bit0: out bf16, bit1: exact GELU.
__global__ void k_gemm(const bf16_t* __restrict__ A, const bf16_t* __restrict__ Bp,
                       const float* __restrict__ bias, void* __restrict__ Out,
                       int M, int N, int K, int flags) {
    const int wave = blockIdx.x * (blockDim.x >> 5) + (threadIdx.x >> 5);
    const int nblk = N >> 6;                         // 4-tile groups along N
    const int total = (M >> 4) * nblk;
    if (wave >= total) return;
    const int tm = wave / nblk;
    const int tb = wave % nblk;
    const int ntiles = N >> 4;

    v8f acc[4] = {};
    const bf16_t* arow = A + (size_t)tm * 16 * K;
    const int kchunks = K >> 5;
    for (int kc = 0; kc < kchunks; ++kc) {
        if (kc + 1 < kchunks) {
            __builtin_prefetch(arow + (kc + 1) * 32, 0, 0);   // global_prefetch_b8
            __builtin_prefetch(Bp + ((size_t)(kc + 1) * ntiles + tb * 4) * 512, 0, 0);
        }
        const v16bf af = load_a_rowmajor(arow + kc * 32, K);
#pragma unroll
        for (int t = 0; t < 4; ++t) {
            const int tn = tb * 4 + t;
            const v16bf bfr = load_b_packed(Bp + ((size_t)kc * ntiles + tn) * 512);
            acc[t] = wmma_bf16(af, bfr, acc[t]);
        }
    }
    const int l     = lane_id();
    const int mbase = tm * 16 + ((l >> 4) << 3);
#pragma unroll
    for (int t = 0; t < 4; ++t) {
        const int ncol = (tb * 4 + t) * 16 + (l & 15);
        const float bv = bias[ncol];
#pragma unroll
        for (int r = 0; r < 8; ++r) {
            float v = acc[t][r] + bv;
            if (flags & 2) v = 0.5f * v * (1.0f + erff(v * 0.70710678118654752f));
            const size_t o = (size_t)(mbase + r) * N + ncol;
            if (flags & 1) ((bf16_t*)Out)[o] = (bf16_t)v;
            else           ((float*)Out)[o]  = v;
        }
    }
}

// ---------------------------------------------------------------------------
// Windowed attention: one block per (window, head). 4 waves x 16 query rows.
__global__ void __launch_bounds__(128)
k_win_attn(const bf16_t* __restrict__ qkv, const float* __restrict__ rpb,
           const int* __restrict__ ridx, bf16_t* __restrict__ out) {
    const int win = blockIdx.x / HEADS_;
    const int h   = blockIdx.x % HEADS_;
    __shared__ bf16_t Ks[NL_ * HD_];      // [key][hd]   (hd contiguous)
    __shared__ bf16_t Vst[HD_ * NL_];     // [hd][key]   (key contiguous, transposed)
    __shared__ bf16_t Ps[4 * 16 * NL_];

    const int tid = threadIdx.x;
    const size_t base = (size_t)win * NL_;
    // K: 16B vector copies; V: scalar transpose into Vst
    for (int i = tid; i < NL_ * 4; i += 128) {
        const int row = i >> 2, seg = i & 3;
        *(v8bf*)(Ks + row * HD_ + seg * 8) =
            *(const v8bf*)(qkv + (base + row) * 576 + 192 + h * HD_ + seg * 8);
    }
    for (int i = tid; i < NL_ * HD_; i += 128) {
        const int row = i >> 5, d = i & 31;
        Vst[d * NL_ + row] = qkv[(base + row) * 576 + 384 + h * HD_ + d];
    }
    __syncthreads();

    const int wv = tid >> 5;
    const int l  = tid & 31;
    const int mrow = ((l >> 4) << 3);
    const v16bf qf = load_a_rowmajor(qkv + (base + wv * 16) * 576 + h * HD_, 576);

    v8f sacc[4];
#pragma unroll
    for (int kt = 0; kt < 4; ++kt) {
        v8f z = {};
        sacc[kt] = wmma_bf16(qf, load_b_contig(Ks + kt * 16 * HD_, HD_), z);
    }

    float sv[4][8];
#pragma unroll
    for (int kt = 0; kt < 4; ++kt)
#pragma unroll
        for (int r = 0; r < 8; ++r) {
            const int nq = wv * 16 + mrow + r;
            const int nk = kt * 16 + (l & 15);
            sv[kt][r] = sacc[kt][r] * SCALE_ + rpb[ridx[nq * NL_ + nk] * HEADS_ + h];
        }

#pragma unroll
    for (int r = 0; r < 8; ++r) {
        float m = -1e30f;
#pragma unroll
        for (int kt = 0; kt < 4; ++kt) m = fmaxf(m, sv[kt][r]);
#pragma unroll
        for (int d = 8; d >= 1; d >>= 1) m = fmaxf(m, __shfl_xor(m, d, 32));
        float s = 0.f;
#pragma unroll
        for (int kt = 0; kt < 4; ++kt) { sv[kt][r] = __expf(sv[kt][r] - m); s += sv[kt][r]; }
#pragma unroll
        for (int d = 8; d >= 1; d >>= 1) s += __shfl_xor(s, d, 32);
        const float rinv = 1.0f / s;
#pragma unroll
        for (int kt = 0; kt < 4; ++kt)
            Ps[(wv * 16 + mrow + r) * NL_ + kt * 16 + (l & 15)] = (bf16_t)(sv[kt][r] * rinv);
    }
    __syncthreads();

    // out(16x32) = P(16x64) @ V(64x32), V read transposed-contiguous
#pragma unroll
    for (int nt = 0; nt < 2; ++nt) {
        v8f oa = {};
#pragma unroll
        for (int kc = 0; kc < 2; ++kc)
            oa = wmma_bf16(load_a_rowmajor(Ps + wv * 16 * NL_ + kc * 32, NL_),
                           load_b_contig(Vst + (nt * 16) * NL_ + kc * 32, NL_), oa);
#pragma unroll
        for (int r = 0; r < 8; ++r)
            out[(base + wv * 16 + mrow + r) * C_ + h * HD_ + nt * 16 + (l & 15)] = (bf16_t)oa[r];
    }
}

// ---------------------------------------------------------------------------
// Global attention: flash-style online softmax over 2048 keys.
__global__ void __launch_bounds__(128)
k_glob_attn(const bf16_t* __restrict__ qkv, const float* __restrict__ rpb,
            const int* __restrict__ ridx, bf16_t* __restrict__ out) {
    const int qt = blockIdx.x & 31;
    const int h  = (blockIdx.x >> 5) % HEADS_;
    const int b  = blockIdx.x / (32 * HEADS_);
    __shared__ bf16_t Ks[64 * HD_];
    __shared__ bf16_t Vst[HD_ * 64];
    __shared__ bf16_t Ps[4 * 16 * 64];

    const int tid = threadIdx.x;
    const int wv  = tid >> 5;
    const int l   = tid & 31;
    const int mrow = ((l >> 4) << 3);
    const size_t gbase = (size_t)ROWS_LOCAL_ + (size_t)b * NG_;
    const int q0 = qt * 64 + wv * 16;

    const v16bf qf = load_a_rowmajor(qkv + (gbase + q0) * 576 + h * HD_, 576);

    v8f oacc[2] = {};
    float m_i[8], l_i[8];
#pragma unroll
    for (int r = 0; r < 8; ++r) { m_i[r] = -1e30f; l_i[r] = 0.f; }

    for (int kt0 = 0; kt0 < NG_; kt0 += 64) {
        __syncthreads();
        for (int i = tid; i < 64 * 4; i += 128) {
            const int row = i >> 2, seg = i & 3;
            *(v8bf*)(Ks + row * HD_ + seg * 8) =
                *(const v8bf*)(qkv + (gbase + kt0 + row) * 576 + 192 + h * HD_ + seg * 8);
        }
        for (int i = tid; i < 64 * HD_; i += 128) {
            const int row = i >> 5, d = i & 31;
            Vst[d * 64 + row] = qkv[(gbase + kt0 + row) * 576 + 384 + h * HD_ + d];
        }
        __syncthreads();

        v8f sacc[4];
#pragma unroll
        for (int kt = 0; kt < 4; ++kt) {
            v8f z = {};
            sacc[kt] = wmma_bf16(qf, load_b_contig(Ks + kt * 16 * HD_, HD_), z);
        }
        float sv[4][8];
#pragma unroll
        for (int kt = 0; kt < 4; ++kt)
#pragma unroll
            for (int r = 0; r < 8; ++r) {
                const int nq = q0 + mrow + r;
                const int nk = kt0 + kt * 16 + (l & 15);
                sv[kt][r] = sacc[kt][r] * SCALE_
                          + rpb[ridx[(size_t)nq * NG_ + nk] * HEADS_ + h];
            }
#pragma unroll
        for (int r = 0; r < 8; ++r) {
            float rm = -1e30f;
#pragma unroll
            for (int kt = 0; kt < 4; ++kt) rm = fmaxf(rm, sv[kt][r]);
#pragma unroll
            for (int d = 8; d >= 1; d >>= 1) rm = fmaxf(rm, __shfl_xor(rm, d, 32));
            const float newm  = fmaxf(m_i[r], rm);
            const float alpha = __expf(m_i[r] - newm);
            float s = 0.f;
#pragma unroll
            for (int kt = 0; kt < 4; ++kt) { sv[kt][r] = __expf(sv[kt][r] - newm); s += sv[kt][r]; }
#pragma unroll
            for (int d = 8; d >= 1; d >>= 1) s += __shfl_xor(s, d, 32);
            l_i[r] = l_i[r] * alpha + s;
            m_i[r] = newm;
            oacc[0][r] *= alpha;
            oacc[1][r] *= alpha;
#pragma unroll
            for (int kt = 0; kt < 4; ++kt)
                Ps[(wv * 16 + mrow + r) * 64 + kt * 16 + (l & 15)] = (bf16_t)sv[kt][r];
        }
#pragma unroll
        for (int nt = 0; nt < 2; ++nt)
#pragma unroll
            for (int kc = 0; kc < 2; ++kc)
                oacc[nt] = wmma_bf16(load_a_rowmajor(Ps + wv * 16 * 64 + kc * 32, 64),
                                     load_b_contig(Vst + (nt * 16) * 64 + kc * 32, 64),
                                     oacc[nt]);
    }
#pragma unroll
    for (int nt = 0; nt < 2; ++nt)
#pragma unroll
        for (int r = 0; r < 8; ++r)
            out[(gbase + q0 + mrow + r) * C_ + h * HD_ + nt * 16 + (l & 15)]
                = (bf16_t)(oacc[nt][r] / l_i[r]);
}

// ---------------------------------------------------------------------------
__global__ void k_resid_local(const float* __restrict__ x0, const float* __restrict__ proj,
                              float* __restrict__ res2) {
    const size_t i = (size_t)blockIdx.x * blockDim.x + threadIdx.x;
    if (i >= (size_t)ROWS_LOCAL_ * C_) return;
    const size_t row = i / C_;
    const int c = (int)(i % C_);
    const int b  = (int)(row >> 17);
    const int wr = (int)(row & 63);
    const int wn = (int)((row >> 6) & (NWB_ - 1));
    const int sI = ((wn >> 8) << 2) + (wr >> 4);
    const int hI = (((wn >> 4) & 15) << 2) + ((wr >> 2) & 3);
    const int wI = ((wn & 15) << 2) + (wr & 3);
    const size_t tok = (size_t)b * L_ + (size_t)sI * 4096 + hI * 64 + wI;
    res2[i] = proj[i] + x0[tok * C_ + c];
}

__global__ void k_resid_global(const float* __restrict__ gx0, const float* __restrict__ proj,
                               float* __restrict__ res2) {
    const size_t i = (size_t)blockIdx.x * blockDim.x + threadIdx.x;
    if (i >= (size_t)ROWS_GLOB_ * C_) return;
    const size_t o = (size_t)ROWS_LOCAL_ * C_ + i;
    res2[o] = proj[o] + gx0[i];
}

__global__ void k_fuse(float* __restrict__ res2, const float* __restrict__ l2g_w,
                       const float* __restrict__ l2g_b) {
    const int wn = blockIdx.x;
    const int c  = threadIdx.x;
    const size_t grow = ((size_t)ROWS_LOCAL_ + wn) * C_ + c;
    const float g = res2[grow];
    const size_t base = (size_t)wn * NL_ * C_;
    float s = 0.f;
#pragma unroll 4
    for (int w = 0; w < NL_; ++w) {
        const float v = res2[base + (size_t)w * C_ + c] + g;
        res2[base + (size_t)w * C_ + c] = v;
        s += v * l2g_w[w];
    }
    res2[grow] = g + s + l2g_b[0];
}

__global__ void k_final(const float* __restrict__ res2, const float* __restrict__ mlp,
                        float* __restrict__ out) {
    const size_t i = (size_t)blockIdx.x * blockDim.x + threadIdx.x;
    if (i >= (size_t)B_ * (L_ + NG_) * C_) return;
    const size_t row = i / C_;
    const int c = (int)(i % C_);
    const int b = (int)(row / (L_ + NG_));
    const int t = (int)(row % (L_ + NG_));
    size_t srow;
    if (t < L_) {
        const int sI = t >> 12, hI = (t >> 6) & 63, wI = t & 63;
        const int win = ((sI >> 2) << 8) + ((hI >> 2) << 4) + (wI >> 2);
        const int wr  = ((sI & 3) << 4) + ((hI & 3) << 2) + (wI & 3);
        srow = ((size_t)b * NWB_ + win) * NL_ + wr;
    } else {
        srow = (size_t)ROWS_LOCAL_ + (size_t)b * NG_ + (t - L_);
    }
    out[i] = res2[srow * C_ + c] + mlp[srow * C_ + c];
}

// ---------------------------------------------------------------------------
extern "C" void kernel_launch(void* const* d_in, const int* in_sizes, int n_in,
                              void* d_out, int out_size, void* d_ws, size_t ws_size,
                              hipStream_t stream) {
    (void)in_sizes; (void)n_in; (void)out_size; (void)ws_size;
    const float* x       = (const float*)d_in[0];
    const float* gx      = (const float*)d_in[1];
    const float* qkv_w   = (const float*)d_in[2];
    const float* qkv_b   = (const float*)d_in[3];
    const float* proj_w  = (const float*)d_in[4];
    const float* proj_b  = (const float*)d_in[5];
    const float* rpb_l   = (const float*)d_in[6];
    const float* rpb_g   = (const float*)d_in[7];
    const float* n1g     = (const float*)d_in[8];
    const float* n1b     = (const float*)d_in[9];
    const float* n2g     = (const float*)d_in[10];
    const float* n2b     = (const float*)d_in[11];
    const float* fc1_w   = (const float*)d_in[12];
    const float* fc1_b   = (const float*)d_in[13];
    const float* fc2_w   = (const float*)d_in[14];
    const float* fc2_b   = (const float*)d_in[15];
    const float* l2g_w   = (const float*)d_in[16];
    const float* l2g_b   = (const float*)d_in[17];
    const int*   ridx_l  = (const int*)d_in[18];
    const int*   ridx_g  = (const int*)d_in[19];
    float* out = (float*)d_out;

    char* ws = (char*)d_ws;
    size_t off = 0;
    auto alloc = [&](size_t bytes) { void* p = ws + off; off += (bytes + 255) & ~(size_t)255; return p; };
    bf16_t* wq  = (bf16_t*)alloc((size_t)C_ * 576 * 2);          // packed
    bf16_t* wp  = (bf16_t*)alloc((size_t)C_ * C_ * 2);           // packed
    bf16_t* w1  = (bf16_t*)alloc((size_t)C_ * HID_ * 2);         // packed
    bf16_t* w2  = (bf16_t*)alloc((size_t)HID_ * C_ * 2);         // packed
    bf16_t* xn  = (bf16_t*)alloc((size_t)ROWS_ALL_ * C_ * 2);    // LN1 out, later LN2 out
    bf16_t* qkv = (bf16_t*)alloc((size_t)ROWS_ALL_ * HID_ * 2);  // qkv (576), later hidden (768)
    bf16_t* attn= (bf16_t*)alloc((size_t)ROWS_ALL_ * C_ * 2);
    float*  pout= (float*)alloc((size_t)ROWS_ALL_ * C_ * 4);     // proj out, later fc2 out
    float*  res2= (float*)alloc((size_t)ROWS_ALL_ * C_ * 4);

    // 1) pack weights into bf16 WMMA fragment order
    k_pack_b<<<(C_ * 576 + 255) / 256, 256, 0, stream>>>(qkv_w, wq, C_, 576);
    k_pack_b<<<(C_ * C_ + 255) / 256, 256, 0, stream>>>(proj_w, wp, C_, C_);
    k_pack_b<<<(C_ * HID_ + 255) / 256, 256, 0, stream>>>(fc1_w, w1, C_, HID_);
    k_pack_b<<<(HID_ * C_ + 255) / 256, 256, 0, stream>>>(fc2_w, w2, HID_, C_);

    // 2) LN1
    k_ln<<<ROWS_LOCAL_ / 8, 256, 0, stream>>>(x, n1g, n1b, xn, ROWS_LOCAL_, 1, 0);
    k_ln<<<ROWS_GLOB_ / 8, 256, 0, stream>>>(gx, n1g, n1b, xn, ROWS_GLOB_, 0, ROWS_LOCAL_);

    // 3) QKV GEMM (266240 x 576 x 192)
    {
        const int waves = (ROWS_ALL_ / 16) * (576 / 64);
        k_gemm<<<(waves + 7) / 8, 256, 0, stream>>>(xn, wq, qkv_b, qkv,
                                                    ROWS_ALL_, 576, C_, 1);
    }

    // 4) attention
    k_win_attn<<<(B_ * NWB_) * HEADS_, 128, 0, stream>>>(qkv, rpb_l, ridx_l, attn);
    k_glob_attn<<<B_ * HEADS_ * 32, 128, 0, stream>>>(qkv, rpb_g, ridx_g, attn);

    // 5) proj GEMM -> f32
    {
        const int waves = (ROWS_ALL_ / 16) * (C_ / 64);
        k_gemm<<<(waves + 7) / 8, 256, 0, stream>>>(attn, wp, proj_b, pout,
                                                    ROWS_ALL_, C_, C_, 0);
    }

    // 6) residuals + local<->global fuse
    k_resid_local<<<((size_t)ROWS_LOCAL_ * C_ + 255) / 256, 256, 0, stream>>>(x, pout, res2);
    k_resid_global<<<((size_t)ROWS_GLOB_ * C_ + 255) / 256, 256, 0, stream>>>(gx, pout, res2);
    k_fuse<<<B_ * NWB_, C_, 0, stream>>>(res2, l2g_w, l2g_b);

    // 7) LN2 -> xn (reuse)
    k_ln<<<ROWS_ALL_ / 8, 256, 0, stream>>>(res2, n2g, n2b, xn, ROWS_ALL_, 0, 0);

    // 8) MLP
    {
        const int waves = (ROWS_ALL_ / 16) * (HID_ / 64);
        k_gemm<<<(waves + 7) / 8, 256, 0, stream>>>(xn, w1, fc1_b, qkv,
                                                    ROWS_ALL_, HID_, C_, 1 | 2);
    }
    {
        const int waves = (ROWS_ALL_ / 16) * (C_ / 64);
        k_gemm<<<(waves + 7) / 8, 256, 0, stream>>>(qkv, w2, fc2_b, pout,
                                                    ROWS_ALL_, C_, HID_, 0);
    }

    // 9) residual + window-reverse + concat
    k_final<<<((size_t)B_ * (L_ + NG_) * C_ + 255) / 256, 256, 0, stream>>>(res2, pout, out);
}
